// PolyAttention_72739566125493
// MI455X (gfx1250) — matrix operand order repaired and checked by reading, not compiled, not run on hardware
//
#include <hip/hip_runtime.h>

#define BATCH 2
#define SEQ   2048
#define DM    2048
#define NH    16
#define HD    128

typedef __attribute__((ext_vector_type(16))) __bf16 v16bf;
typedef __attribute__((ext_vector_type(8)))  float  v8f;
typedef __attribute__((ext_vector_type(4)))  unsigned int u32x4;
typedef __attribute__((ext_vector_type(8)))  int          i32x8;
typedef __attribute__((ext_vector_type(4)))  int          i32x4;

union Frag { uint4 u[2]; v16bf v; };

#if defined(__gfx1250__) && __has_builtin(__builtin_amdgcn_tensor_load_to_lds) && \
    __has_builtin(__builtin_amdgcn_s_wait_tensorcnt)
#define USE_TDM 1
#else
#define USE_TDM 0
#endif

__device__ __forceinline__ v8f wmma_bf16(v16bf a, v16bf b, v8f c) {
  // D = A(16x32 bf16) * B(32x16 bf16) + C(16x16 f32)
  return __builtin_amdgcn_wmma_f32_16x16x32_bf16(false, a, false, b, (short)0, c,
                                                 false, false);
}

__device__ __forceinline__ v8f zero8() {
  v8f z = {0.f, 0.f, 0.f, 0.f, 0.f, 0.f, 0.f, 0.f};
  return z;
}

#if USE_TDM
// Tensor Data Mover: DMA a 128-row x 32-col bf16 tile (row stride = DM elems)
// from global to LDS. D# built per ISA 8.3/8.4:
//  group0: count=1 | lds_addr | global_addr[56:0] | type=2
//  group1: data_size=2B, tensor_dim0=DM, tensor_dim1=BATCH*SEQ,
//          tile_dim0=32, tile_dim1=128, tensor_dim0_stride=DM
// This toolchain exposes the 6-arg builtin:
//  (u32x4 g0, i32x8 g1, i32x4 g2, i32x4 g3, i32x8 pad, i32 cpol)
__device__ __forceinline__ void tdm_load_tile_128x32(const __bf16* gsrc,
                                                     unsigned lds_byte) {
  unsigned long long ga = (unsigned long long)(uintptr_t)gsrc;
  u32x4 g0;
  g0[0] = 1u;                                           // count = 1 (valid)
  g0[1] = lds_byte;                                     // LDS byte address
  g0[2] = (unsigned)ga;                                 // global addr lo
  g0[3] = (unsigned)((ga >> 32) & 0x01FFFFFFull) | (2u << 30);  // hi | type=2
  i32x8 g1;
  g1[0] = 0x00010000;                                   // data_size = 2 bytes
  g1[1] = (int)(((unsigned)DM & 0xFFFFu) << 16);        // tensor_dim0[15:0]
  g1[2] = (int)(((unsigned)DM >> 16) |
                (((unsigned)(BATCH * SEQ) & 0xFFFFu) << 16));  // td0 hi | td1 lo
  g1[3] = (int)(((unsigned)(BATCH * SEQ) >> 16) | (32u << 16)); // td1 hi | tile_dim0
  g1[4] = 128;                                          // tile_dim1 (tile_dim2=0)
  g1[5] = DM;                                           // tensor_dim0_stride lo32
  g1[6] = 0;
  g1[7] = 0;
  i32x4 gz4 = {0, 0, 0, 0};                             // 2D tile: groups 2/3 zero
  i32x8 gz8 = {0, 0, 0, 0, 0, 0, 0, 0};
  __builtin_amdgcn_tensor_load_to_lds(g0, g1, gz4, gz4, gz8, 0);
}
#endif

// ---------------------------------------------------------------------------
// fp32 -> bf16 conversion
// ---------------------------------------------------------------------------
__global__ void cvt_bf16_kernel(const float* __restrict__ src,
                                __bf16* __restrict__ dst, int n) {
  int i = blockIdx.x * blockDim.x + threadIdx.x;
  if (i < n) dst[i] = (__bf16)src[i];
}

// ---------------------------------------------------------------------------
// C[M,N] = A[M,K] * B[N,K]^T  (both row-major along K), bf16 in, f32 accum.
// 256 threads = 8 wave32, 128x128 block tile, K-step 32, ping-pong LDS.
// TDM path: wave 0 DMAs both tiles; others gate on tensorcnt+barrier.
// mode 0: bf16 -> [b][h][t][hd]   (Q, K pre-RoPE)
// mode 1: bf16 -> [b][h][hd][t]   (V transposed for the PV WMMA B-operand)
// mode 2: f32  -> [row][col]      (output projection)
// ---------------------------------------------------------------------------
__global__ __launch_bounds__(256)
void gemm_bf16_kernel(const __bf16* __restrict__ A, const __bf16* __restrict__ B,
                      __bf16* __restrict__ dstb, float* __restrict__ dstf,
                      int M, int N, int K, int mode) {
  __shared__ __align__(32) __bf16 As[2][128][32];
  __shared__ __align__(32) __bf16 Bs[2][128][32];
  (void)M;

  const int tid  = threadIdx.x;
  const int lane = tid & 31;
  const int r    = lane & 15;
  const int hi   = lane >> 4;
  const int wid  = tid >> 5;
  const int wm   = (wid & 3) * 32;
  const int wn   = (wid >> 2) * 64;
  const int bm   = blockIdx.y * 128;
  const int bn   = blockIdx.x * 128;

  v8f acc[2][4];
#pragma unroll
  for (int i = 0; i < 2; ++i)
#pragma unroll
    for (int j = 0; j < 4; ++j) acc[i][j] = zero8();

  auto compute_stage = [&](int st) {
    Frag af[2], bfr[4];
#pragma unroll
    for (int mt = 0; mt < 2; ++mt) {
      // A layout: lane = row r; K chunks [hi*8..+7] and [16+hi*8..+7]
      af[mt].u[0] = *(const uint4*)&As[st][wm + mt * 16 + r][hi * 8];
      af[mt].u[1] = *(const uint4*)&As[st][wm + mt * 16 + r][16 + hi * 8];
    }
#pragma unroll
    for (int nt = 0; nt < 4; ++nt) {
      // B layout: lane = column r; K chunks [hi*8..+7] and [16+hi*8..+7]
      bfr[nt].u[0] = *(const uint4*)&Bs[st][wn + nt * 16 + r][hi * 8];
      bfr[nt].u[1] = *(const uint4*)&Bs[st][wn + nt * 16 + r][16 + hi * 8];
    }
#pragma unroll
    for (int mt = 0; mt < 2; ++mt)
#pragma unroll
      for (int nt = 0; nt < 4; ++nt)
        acc[mt][nt] = wmma_bf16(af[mt].v, bfr[nt].v, acc[mt][nt]);
  };

#if USE_TDM
  const __bf16* gA0 = A + (size_t)bm * K;
  const __bf16* gB0 = B + (size_t)bn * K;
  if (wid == 0) {
    tdm_load_tile_128x32(gA0, (unsigned)(uintptr_t)&As[0][0][0]);
    tdm_load_tile_128x32(gB0, (unsigned)(uintptr_t)&Bs[0][0][0]);
  }
  for (int k0 = 0; k0 < K; k0 += 32) {
    const int st = (k0 >> 5) & 1;
    const bool more = (k0 + 32) < K;
    if (wid == 0) {
      if (more) {  // prefetch next tile via TDM while this one is consumed
        tdm_load_tile_128x32(gA0 + k0 + 32, (unsigned)(uintptr_t)&As[st ^ 1][0][0]);
        tdm_load_tile_128x32(gB0 + k0 + 32, (unsigned)(uintptr_t)&Bs[st ^ 1][0][0]);
        __builtin_amdgcn_s_wait_tensorcnt(2);  // stage-st pair (older) complete
      } else {
        __builtin_amdgcn_s_wait_tensorcnt(0);
      }
    }
    __syncthreads();     // stage st visible to all waves
    compute_stage(st);
    __syncthreads();     // all reads of st done before TDM overwrites it
  }
#else
  const int lrow = tid >> 1;          // 0..127
  const int lcol = (tid & 1) * 16;    // 0 or 16
  const __bf16* gArow = A + (size_t)(bm + lrow) * K + lcol;
  const __bf16* gBrow = B + (size_t)(bn + lrow) * K + lcol;

  {  // prologue: fill stage 0
    uint4 a0 = *(const uint4*)(gArow);
    uint4 a1 = *(const uint4*)(gArow + 8);
    uint4 b0 = *(const uint4*)(gBrow);
    uint4 b1 = *(const uint4*)(gBrow + 8);
    *(uint4*)&As[0][lrow][lcol]     = a0;
    *(uint4*)&As[0][lrow][lcol + 8] = a1;
    *(uint4*)&Bs[0][lrow][lcol]     = b0;
    *(uint4*)&Bs[0][lrow][lcol + 8] = b1;
  }
  for (int k0 = 0; k0 < K; k0 += 32) {
    const int st = (k0 >> 5) & 1;
    const bool more = (k0 + 32) < K;
    uint4 na0, na1, nb0, nb1;
    if (more) {  // issue next-tile global loads early; overlap with WMMAs
      __builtin_prefetch(gArow + k0 + 64, 0, 1);
      __builtin_prefetch(gBrow + k0 + 64, 0, 1);
      na0 = *(const uint4*)(gArow + k0 + 32);
      na1 = *(const uint4*)(gArow + k0 + 40);
      nb0 = *(const uint4*)(gBrow + k0 + 32);
      nb1 = *(const uint4*)(gBrow + k0 + 40);
    }
    __syncthreads();     // stage st stores (prev iter) visible
    compute_stage(st);
    if (more) {
      *(uint4*)&As[st ^ 1][lrow][lcol]     = na0;
      *(uint4*)&As[st ^ 1][lrow][lcol + 8] = na1;
      *(uint4*)&Bs[st ^ 1][lrow][lcol]     = nb0;
      *(uint4*)&Bs[st ^ 1][lrow][lcol + 8] = nb1;
    }
  }
#endif

  // Epilogue. C layout: lane -> N = r, VGPR rr -> M = rr + hi*8.
#pragma unroll
  for (int mt = 0; mt < 2; ++mt)
#pragma unroll
    for (int nt = 0; nt < 4; ++nt)
#pragma unroll
      for (int rr = 0; rr < 8; ++rr) {
        int row = bm + wm + mt * 16 + rr + hi * 8;
        int col = bn + wn + nt * 16 + r;
        float val = acc[mt][nt][rr];
        if (mode == 2) {
          dstf[(size_t)row * N + col] = val;
        } else {
          int bb = row / SEQ, t = row % SEQ;
          int h = col >> 7, hd = col & 127;
          size_t idx = (mode == 0)
                         ? ((((size_t)bb * NH + h) * SEQ + t) * HD + hd)
                         : ((((size_t)bb * NH + h) * HD + hd) * SEQ + t);
          dstb[idx] = (__bf16)val;
        }
      }
}

// ---------------------------------------------------------------------------
// In-place RoPE on [b][h][t][hd] bf16 tensor. One thread per (b,h,t,j<64).
// ---------------------------------------------------------------------------
__global__ void rope_kernel(__bf16* __restrict__ q) {
  int idx = blockIdx.x * blockDim.x + threadIdx.x;  // total = B*H*SEQ*64 = 2^22
  int j  = idx & 63;
  int t  = (idx >> 6) & (SEQ - 1);
  int bh = idx >> 17;
  size_t base = ((size_t)bh * SEQ + t) * HD;
  float theta = __expf(-(float)j * (__logf(10000.0f) / 64.0f));
  float ang = (float)t * theta;
  float c, s;
  __sincosf(ang, &s, &c);
  float x1 = (float)q[base + j];
  float x2 = (float)q[base + j + 64];
  q[base + j]      = (__bf16)(x1 * c - x2 * s);
  q[base + j + 64] = (__bf16)(x1 * s + x2 * c);
}

// ---------------------------------------------------------------------------
// Poly attention: 8 independent wave32 per block, one 16-query tile each.
// scores = (q.k)^4, causal, denom via WMMA against all-ones B (row sums land
// in the same lane/VGPR mapping as the output), out = (S@V)/denom.
// Per-wave LDS slice for the S C->A relayout; LDS is in-order per wave, so
// no barriers are needed.
// ---------------------------------------------------------------------------
__global__ __launch_bounds__(256)
void poly_attn_kernel(const __bf16* __restrict__ qb, const __bf16* __restrict__ kb,
                      const __bf16* __restrict__ vt, __bf16* __restrict__ ab) {
  __shared__ __align__(32) __bf16 sS[8][16][32];

  const int wid  = threadIdx.x >> 5;
  const int lane = threadIdx.x & 31;
  const int r  = lane & 15;
  const int hi = lane >> 4;
  const int qt = blockIdx.x * 8 + wid;
  const int h = blockIdx.y, b = blockIdx.z;
  const int m0 = qt * 16;

  const size_t bh  = (size_t)b * NH + h;
  const __bf16* Q  = qb + bh * SEQ * HD;
  const __bf16* Km = kb + bh * SEQ * HD;
  const __bf16* Vt = vt + bh * HD * SEQ;

  Frag qa[4];
#pragma unroll
  for (int kc = 0; kc < 4; ++kc) {
    const __bf16* row = Q + (size_t)(m0 + r) * HD + kc * 32;
    qa[kc].u[0] = *(const uint4*)(row + hi * 8);
    qa[kc].u[1] = *(const uint4*)(row + 16 + hi * 8);
  }

  v16bf ones;
#pragma unroll
  for (int i = 0; i < 16; ++i) ones[i] = (__bf16)1.0f;

  v8f acc_o[8];
#pragma unroll
  for (int i = 0; i < 8; ++i) acc_o[i] = zero8();
  v8f acc_d = zero8();

  const int kend = m0 + 16;  // causal bound
  for (int key0 = 0; key0 < kend; key0 += 32) {
#pragma unroll
    for (int g = 0; g < 2; ++g) {
      v8f s = zero8();
      const int kn = key0 + g * 16 + r;
      const __bf16* krow = Km + (size_t)kn * HD;
      Frag kf;
#pragma unroll
      for (int kc = 0; kc < 4; ++kc) {
        kf.u[0] = *(const uint4*)(krow + kc * 32 + hi * 8);
        kf.u[1] = *(const uint4*)(krow + kc * 32 + 16 + hi * 8);
        s = wmma_bf16(qa[kc].v, kf.v, s);
      }
#pragma unroll
      for (int rr = 0; rr < 8; ++rr) {
        int m = m0 + rr + hi * 8;
        float v = s[rr];
        float v2 = v * v;
        float v4 = v2 * v2;
        v4 = (kn <= m) ? v4 : 0.0f;  // mask after even power, as in reference
        sS[wid][rr + hi * 8][g * 16 + r] = (__bf16)v4;
      }
    }
    // Reload S as A-fragment (per-wave LDS in-order: no barrier needed)
    Frag sf;
    sf.u[0] = *(const uint4*)&sS[wid][r][hi * 8];
    sf.u[1] = *(const uint4*)&sS[wid][r][16 + hi * 8];

    acc_d = wmma_bf16(sf.v, ones, acc_d);  // rowsum denominator

#pragma unroll
    for (int nd = 0; nd < 8; ++nd) {
      const __bf16* vrow = Vt + (size_t)(nd * 16 + r) * SEQ + key0;
      Frag vf;
      vf.u[0] = *(const uint4*)(vrow + hi * 8);
      vf.u[1] = *(const uint4*)(vrow + 16 + hi * 8);
      acc_o[nd] = wmma_bf16(sf.v, vf.v, acc_o[nd]);
    }
  }

#pragma unroll
  for (int nd = 0; nd < 8; ++nd)
#pragma unroll
    for (int rr = 0; rr < 8; ++rr) {
      int m = m0 + rr + hi * 8;
      float den = acc_d[rr];
      den = den > 1e-6f ? den : 1e-6f;
      float val = acc_o[nd][rr] / den;
      ab[((size_t)b * SEQ + m) * DM + h * HD + nd * 16 + r] = (__bf16)val;
    }
}

// ---------------------------------------------------------------------------
extern "C" void kernel_launch(void* const* d_in, const int* in_sizes, int n_in,
                              void* d_out, int out_size, void* d_ws, size_t ws_size,
                              hipStream_t stream) {
  (void)in_sizes; (void)n_in; (void)out_size; (void)ws_size;
  const float* x  = (const float*)d_in[0];
  const float* Wq = (const float*)d_in[1];
  const float* Wk = (const float*)d_in[2];
  const float* Wv = (const float*)d_in[3];
  const float* Wo = (const float*)d_in[4];
  float* out = (float*)d_out;

  const size_t XN = (size_t)BATCH * SEQ * DM;  // 8388608
  const size_t WN = (size_t)DM * DM;           // 4194304
  __bf16* ws = (__bf16*)d_ws;                  // ~112 MB total
  __bf16* xb = ws;
  __bf16* wq = xb + XN;
  __bf16* wk = wq + WN;
  __bf16* wv = wk + WN;
  __bf16* wo = wv + WN;
  __bf16* qb = wo + WN;   // [b][h][t][hd]
  __bf16* kb = qb + XN;   // [b][h][t][hd]
  __bf16* vt = kb + XN;   // [b][h][hd][t]
  __bf16* ab = vt + XN;   // [b][t][d]

  cvt_bf16_kernel<<<(XN + 255) / 256, 256, 0, stream>>>(x,  xb, (int)XN);
  cvt_bf16_kernel<<<(WN + 255) / 256, 256, 0, stream>>>(Wq, wq, (int)WN);
  cvt_bf16_kernel<<<(WN + 255) / 256, 256, 0, stream>>>(Wk, wk, (int)WN);
  cvt_bf16_kernel<<<(WN + 255) / 256, 256, 0, stream>>>(Wv, wv, (int)WN);
  cvt_bf16_kernel<<<(WN + 255) / 256, 256, 0, stream>>>(Wo, wo, (int)WN);

  dim3 gg(DM / 128, (BATCH * SEQ) / 128);  // 16 x 32 blocks
  gemm_bf16_kernel<<<gg, 256, 0, stream>>>(xb, wq, qb, nullptr, BATCH * SEQ, DM, DM, 0);
  gemm_bf16_kernel<<<gg, 256, 0, stream>>>(xb, wk, kb, nullptr, BATCH * SEQ, DM, DM, 0);
  gemm_bf16_kernel<<<gg, 256, 0, stream>>>(xb, wv, vt, nullptr, BATCH * SEQ, DM, DM, 1);

  const int rope_total = BATCH * NH * SEQ * 64;  // 2^22
  rope_kernel<<<rope_total / 256, 256, 0, stream>>>(qb);
  rope_kernel<<<rope_total / 256, 256, 0, stream>>>(kb);

  poly_attn_kernel<<<dim3(SEQ / 128, NH, BATCH), 256, 0, stream>>>(qb, kb, vt, ab);

  gemm_bf16_kernel<<<gg, 256, 0, stream>>>(ab, wo, nullptr, out, BATCH * SEQ, DM, DM, 2);
}
